// GAT_9586367005317
// MI455X (gfx1250) — matrix operand order repaired and checked by reading, not compiled, not run on hardware
//
#include <hip/hip_runtime.h>

// ---------------------------------------------------------------------------
// GAT layer for MI455X (gfx1250, wave32, WMMA).
// Pipeline:
//   pack W, lin_w -> bf16 hi/lo fragment layout
//   gemm<8>        : h = x @ W              (split-bf16 x3 WMMA, fp32 accuracy)
//   attn           : a_src/a_dst per node   (wave-per-node, shfl reductions)
//   edge_max       : segment max  (ordered-uint atomic max)
//   finalize_m     : decode, where(isfinite, m, 0)
//   edge_exp       : ex = exp(e - m[dst]); denom += ex  (atomic f32 add)
//   edge_agg       : agg[dst] += alpha * h[src]         (wave-per-edge)
//   gemm<4>        : out = relu(agg+bias) @ lin_w + lin_b
// ---------------------------------------------------------------------------

typedef __attribute__((ext_vector_type(16))) __bf16 v16bf;
typedef __attribute__((ext_vector_type(8)))  float  v8f;

#define IN_CH 128
#define HEADS 4
#define CCH   32
#define OUTC  64
#define LRELU 0.2f

// ---- helpers --------------------------------------------------------------

__device__ inline unsigned f2ord(float f) {
    unsigned u = __float_as_uint(f);
    return (u >> 31) ? ~u : (u | 0x80000000u);
}
__device__ inline float ord2f(unsigned u) {
    return __uint_as_float((u >> 31) ? (u & 0x7fffffffu) : ~u);
}

// ---- pack B matrix (K x ncols, fp32) into WMMA B-fragment bf16 hi/lo ------
// Fragment storage: id = ((kt*NT + nt)*32 + lane)*16 + j
//   lane -> g = lane/16, n = lane%16 ; element j -> K_local = 16*g + j
__global__ void pack_b_kernel(const float* __restrict__ src,
                              __bf16* __restrict__ hi, __bf16* __restrict__ lo,
                              int NT, int ncols) {
    int id = blockIdx.x * 256 + threadIdx.x;
    int total = 4 * NT * 512;            // KT=4 (K=128)
    if (id >= total) return;
    int j    = id & 15;
    int lane = (id >> 4) & 31;
    int tile = id >> 9;
    int nt = tile % NT, kt = tile / NT;
    int g = lane >> 4, n = lane & 15;
    int K   = kt * 32 + 16 * g + j;
    int col = nt * 16 + n;
    float f = src[K * ncols + col];
    __bf16 h = (__bf16)f;
    hi[id] = h;
    lo[id] = (__bf16)(f - (float)h);
}

// ---- split-bf16 WMMA GEMM: Out[N x NT*16] = f(A[N x 128]) @ B -------------
// 256 threads = 8 waves; wave w computes rows [blk*128 + w*16, +16).
template <int NT, bool FUSE_IN, bool FUSE_OUT>
__global__ void __launch_bounds__(256)
gat_gemm_kernel(const float* __restrict__ A,
                const __bf16* __restrict__ Bhi, const __bf16* __restrict__ Blo,
                float* __restrict__ Out, int N,
                const float* __restrict__ inBias,
                const float* __restrict__ outBias) {
    __shared__ float lds[128 * 128];
    const int t = threadIdx.x;
    const int rowBlock = blockIdx.x * 128;

    // stage 128 rows x 128 cols through LDS (coalesced float4 loads)
    for (int i = 0; i < 16; ++i) {
        int flat = (i * 256 + t) * 4;
        int row = flat >> 7;
        int col = flat & 127;
        int grow = rowBlock + row;
        int srow = grow < N ? grow : (N - 1);
        float4 v = *(const float4*)(A + (size_t)srow * IN_CH + col);
        float a[4] = {v.x, v.y, v.z, v.w};
        if (FUSE_IN) {
            #pragma unroll
            for (int jj = 0; jj < 4; ++jj) {
                float b = a[jj] + inBias[col + jj];
                a[jj] = b > 0.f ? b : 0.f;
            }
        }
        lds[row * 128 + col + 0] = a[0];
        lds[row * 128 + col + 1] = a[1];
        lds[row * 128 + col + 2] = a[2];
        lds[row * 128 + col + 3] = a[3];
    }
    __syncthreads();

    const int wave = t >> 5, lane = t & 31;
    const int mloc = lane & 15, g = lane >> 4;
    const float* arow = &lds[(wave * 16 + mloc) * 128];

    v8f acc[NT] = {};

    for (int kt = 0; kt < 4; ++kt) {
        // Build A fragment (contiguous k = 16*g + j, matching B packing) + hi/lo split
        v16bf ahi, alo;
        #pragma unroll
        for (int j = 0; j < 16; ++j) {
            int kl = j + 16 * g;
            float f = arow[kt * 32 + kl];
            __bf16 h = (__bf16)f;
            ahi[j] = h;
            alo[j] = (__bf16)(f - (float)h);
        }
        #pragma unroll
        for (int nt = 0; nt < NT; ++nt) {
            const v16bf bhi = *(const v16bf*)(Bhi + ((size_t)((kt * NT + nt) * 32 + lane)) * 16);
            const v16bf blo = *(const v16bf*)(Blo + ((size_t)((kt * NT + nt) * 32 + lane)) * 16);
            acc[nt] = __builtin_amdgcn_wmma_f32_16x16x32_bf16(
                false, ahi, false, bhi, (short)0, acc[nt], false, false);
            acc[nt] = __builtin_amdgcn_wmma_f32_16x16x32_bf16(
                false, ahi, false, blo, (short)0, acc[nt], false, false);
            acc[nt] = __builtin_amdgcn_wmma_f32_16x16x32_bf16(
                false, alo, false, bhi, (short)0, acc[nt], false, false);
        }
    }

    const int ldOut = NT * 16;
    #pragma unroll
    for (int nt = 0; nt < NT; ++nt) {
        #pragma unroll
        for (int r = 0; r < 8; ++r) {
            int row = rowBlock + wave * 16 + r + 8 * g;   // D layout: M = r + 8*g
            if (row < N) {
                int col = nt * 16 + mloc;
                float v = acc[nt][r];
                if (FUSE_OUT) v += outBias[col];
                Out[(size_t)row * ldOut + col] = v;
            }
        }
    }
}

// ---- attention coefficients: wave per node --------------------------------
__global__ void __launch_bounds__(256)
attn_kernel(const float* __restrict__ h, const float* __restrict__ att_src,
            const float* __restrict__ att_dst, float* __restrict__ a_src,
            float* __restrict__ a_dst, int N) {
    int wave = threadIdx.x >> 5, lane = threadIdx.x & 31;
    int node = blockIdx.x * 8 + wave;
    if (node >= N) return;
    float4 hv = *(const float4*)(h + (size_t)node * IN_CH + lane * 4);
    float4 as = *(const float4*)(att_src + lane * 4);
    float4 ad = *(const float4*)(att_dst + lane * 4);
    float s = hv.x * as.x + hv.y * as.y + hv.z * as.z + hv.w * as.w;
    float d = hv.x * ad.x + hv.y * ad.y + hv.z * ad.z + hv.w * ad.w;
    // lanes [8k, 8k+8) cover head k (32 channels); reduce within groups of 8
    #pragma unroll
    for (int off = 1; off < 8; off <<= 1) {
        s += __shfl_xor(s, off, 32);
        d += __shfl_xor(d, off, 32);
    }
    if ((lane & 7) == 0) {
        a_src[node * HEADS + (lane >> 3)] = s;
        a_dst[node * HEADS + (lane >> 3)] = d;
    }
}

// ---- init m / denom / agg -------------------------------------------------
__global__ void init_kernel(unsigned* __restrict__ m, float* __restrict__ denom,
                            float* __restrict__ agg, int N) {
    int i = blockIdx.x * 256 + threadIdx.x;
    if (i < N * IN_CH) agg[i] = 0.f;
    if (i < N * HEADS) { m[i] = 0x007FFFFFu; /* ord(-inf) */ denom[i] = 0.f; }
}

// ---- edge pass 1: segment max ---------------------------------------------
__global__ void edge_max_kernel(const int* __restrict__ src, const int* __restrict__ dst,
                                const float* __restrict__ a_src, const float* __restrict__ a_dst,
                                unsigned* __restrict__ m, int E, int N) {
    int e = blockIdx.x * 256 + threadIdx.x;
    if (e >= E + N) return;
    int s, d;
    if (e < E) { s = src[e]; d = dst[e]; } else { s = d = e - E; }
    float4 asv = *(const float4*)(a_src + (size_t)s * HEADS);
    float4 adv = *(const float4*)(a_dst + (size_t)d * HEADS);
    float av[4] = {asv.x + adv.x, asv.y + adv.y, asv.z + adv.z, asv.w + adv.w};
    #pragma unroll
    for (int hh = 0; hh < HEADS; ++hh) {
        float le = av[hh] > 0.f ? av[hh] : LRELU * av[hh];
        atomicMax(&m[d * HEADS + hh], f2ord(le));
    }
}

__global__ void finalize_m_kernel(unsigned* __restrict__ m, int NH) {
    int i = blockIdx.x * 256 + threadIdx.x;
    if (i >= NH) return;
    float f = ord2f(m[i]);
    if (__builtin_isinf(f) || __builtin_isnan(f)) f = 0.f;
    m[i] = __float_as_uint(f);
}

// ---- edge pass 2: exp + denom ---------------------------------------------
__global__ void edge_exp_kernel(const int* __restrict__ src, const int* __restrict__ dst,
                                const float* __restrict__ a_src, const float* __restrict__ a_dst,
                                const float* __restrict__ m, float* __restrict__ exb,
                                float* __restrict__ denom, int E, int N) {
    int e = blockIdx.x * 256 + threadIdx.x;
    if (e >= E + N) return;
    int s, d;
    if (e < E) { s = src[e]; d = dst[e]; } else { s = d = e - E; }
    float4 asv = *(const float4*)(a_src + (size_t)s * HEADS);
    float4 adv = *(const float4*)(a_dst + (size_t)d * HEADS);
    float4 mv  = *(const float4*)(m + (size_t)d * HEADS);
    float av[4] = {asv.x + adv.x, asv.y + adv.y, asv.z + adv.z, asv.w + adv.w};
    float mm[4] = {mv.x, mv.y, mv.z, mv.w};
    float ex[4];
    #pragma unroll
    for (int hh = 0; hh < HEADS; ++hh) {
        float le = av[hh] > 0.f ? av[hh] : LRELU * av[hh];
        ex[hh] = expf(le - mm[hh]);
        unsafeAtomicAdd(&denom[d * HEADS + hh], ex[hh]);
    }
    *(float4*)(exb + (size_t)e * HEADS) = make_float4(ex[0], ex[1], ex[2], ex[3]);
}

// ---- edge pass 3: weighted scatter-add, wave per edge ---------------------
__global__ void __launch_bounds__(256)
edge_agg_kernel(const int* __restrict__ src, const int* __restrict__ dst,
                const float* __restrict__ exb, const float* __restrict__ denom,
                const float* __restrict__ hbuf, float* __restrict__ agg, int E, int N) {
    int wave = threadIdx.x >> 5, lane = threadIdx.x & 31;
    int e = blockIdx.x * 8 + wave;
    if (e >= E + N) return;
    int s, d;
    if (e < E) { s = src[e]; d = dst[e]; } else { s = d = e - E; }
    float4 exv = *(const float4*)(exb + (size_t)e * HEADS);
    float4 dnv = *(const float4*)(denom + (size_t)d * HEADS);
    float alpha4[4] = {exv.x / (dnv.x + 1e-16f), exv.y / (dnv.y + 1e-16f),
                       exv.z / (dnv.z + 1e-16f), exv.w / (dnv.w + 1e-16f)};
    float alpha = alpha4[lane >> 3];      // channels lane*4..+3 are in head lane/8
    float4 hv = *(const float4*)(hbuf + (size_t)s * IN_CH + lane * 4);
    float* p = agg + (size_t)d * IN_CH + lane * 4;
    unsafeAtomicAdd(p + 0, alpha * hv.x);
    unsafeAtomicAdd(p + 1, alpha * hv.y);
    unsafeAtomicAdd(p + 2, alpha * hv.z);
    unsafeAtomicAdd(p + 3, alpha * hv.w);
}

// ---------------------------------------------------------------------------
extern "C" void kernel_launch(void* const* d_in, const int* in_sizes, int n_in,
                              void* d_out, int out_size, void* d_ws, size_t ws_size,
                              hipStream_t stream) {
    const float* x        = (const float*)d_in[0];
    const int*   eidx     = (const int*)d_in[1];
    const float* W        = (const float*)d_in[2];
    const float* att_src  = (const float*)d_in[3];
    const float* att_dst  = (const float*)d_in[4];
    const float* bias     = (const float*)d_in[5];
    const float* lin_w    = (const float*)d_in[6];
    const float* lin_b    = (const float*)d_in[7];
    float* out = (float*)d_out;

    const int N = in_sizes[0] / IN_CH;
    const int E = in_sizes[1] / 2;
    const int* src = eidx;
    const int* dst = eidx + E;
    const int EP = E + N;   // edges incl. self-loops

    // workspace carve-out (256B aligned regions)
    char* ws = (char*)d_ws;
    size_t off = 0;
    auto carve = [&](size_t bytes) -> void* {
        void* p = ws + off;
        off = (off + bytes + 255) & ~(size_t)255;
        return p;
    };
    __bf16* whi  = (__bf16*)carve((size_t)4 * 8 * 512 * sizeof(__bf16));
    __bf16* wlo  = (__bf16*)carve((size_t)4 * 8 * 512 * sizeof(__bf16));
    __bf16* lwhi = (__bf16*)carve((size_t)4 * 4 * 512 * sizeof(__bf16));
    __bf16* lwlo = (__bf16*)carve((size_t)4 * 4 * 512 * sizeof(__bf16));
    float* h     = (float*)carve((size_t)N * IN_CH * sizeof(float));
    float* a_src = (float*)carve((size_t)N * HEADS * sizeof(float));
    float* a_dst = (float*)carve((size_t)N * HEADS * sizeof(float));
    unsigned* m  = (unsigned*)carve((size_t)N * HEADS * sizeof(unsigned));
    float* denom = (float*)carve((size_t)N * HEADS * sizeof(float));
    float* exb   = (float*)carve((size_t)EP * HEADS * sizeof(float));
    float* agg   = (float*)carve((size_t)N * IN_CH * sizeof(float));
    (void)ws_size; (void)n_in; (void)out_size;

    // 1. pack weights into fragment layout
    pack_b_kernel<<<(4 * 8 * 512 + 255) / 256, 256, 0, stream>>>(W, whi, wlo, 8, 128);
    pack_b_kernel<<<(4 * 4 * 512 + 255) / 256, 256, 0, stream>>>(lin_w, lwhi, lwlo, 4, 64);

    // 2. init accumulators
    init_kernel<<<((size_t)N * IN_CH + 255) / 256, 256, 0, stream>>>(m, denom, agg, N);

    // 3. h = x @ W  (WMMA)
    int gBlocks = (N + 127) / 128;
    gat_gemm_kernel<8, false, false><<<gBlocks, 256, 0, stream>>>(
        x, whi, wlo, h, N, nullptr, nullptr);

    // 4. attention coefficients
    attn_kernel<<<(N + 7) / 8, 256, 0, stream>>>(h, att_src, att_dst, a_src, a_dst, N);

    // 5-7. edge softmax
    edge_max_kernel<<<(EP + 255) / 256, 256, 0, stream>>>(src, dst, a_src, a_dst, m, E, N);
    finalize_m_kernel<<<(N * HEADS + 255) / 256, 256, 0, stream>>>(m, N * HEADS);
    edge_exp_kernel<<<(EP + 255) / 256, 256, 0, stream>>>(
        src, dst, a_src, a_dst, (const float*)m, exb, denom, E, N);

    // 8. weighted aggregation
    edge_agg_kernel<<<(EP + 7) / 8, 256, 0, stream>>>(src, dst, exb, denom, h, agg, E, N);

    // 9. out = relu(agg + bias) @ lin_w + lin_b  (WMMA)
    gat_gemm_kernel<4, true, true><<<gBlocks, 256, 0, stream>>>(
        agg, lwhi, lwlo, out, N, bias, lin_b);
}
